// NRNNAgent_55130200211885
// MI455X (gfx1250) — compile-verified, matrix operand order checked by reading, not run on hardware
//
#include <hip/hip_runtime.h>
#include <stdint.h>

// ---------------------------------------------------------------------------
// MI455X / gfx1250, wave32.
//  - all GEMMs: bf16 operands, f32 accumulate via v_wmma_f32_16x16x32_bf16
//  - tile staging: global_load_async_to_lds_b128 (ASYNCcnt), double buffered
//  - big GEMMs: 128x128 block tile / 8 waves; small batched GEMMs: 64x64 / 4
// ---------------------------------------------------------------------------

typedef __attribute__((ext_vector_type(16))) __bf16 v16bf;
typedef __attribute__((ext_vector_type(8)))  float  v8f;
typedef __attribute__((ext_vector_type(4)))  unsigned int u32x4;
typedef __attribute__((ext_vector_type(2)))  unsigned int u32x2;
typedef __attribute__((ext_vector_type(4)))  float  f32x4;

#define BB   512
#define AA   64
#define EE   256
#define HH   256
#define NAA  32
#define KSEL 19            // ceil((A-1)*0.3)

__device__ __forceinline__ uint16_t f32_to_bf16_rne(float x) {
    uint32_t u = __builtin_bit_cast(uint32_t, x);
    uint32_t r = (u + 0x7FFFu + ((u >> 16) & 1u)) >> 16;
    return (uint16_t)r;
}

union FragBF { v16bf v; u32x4 q[2]; };

__device__ __forceinline__ uint32_t lds_off_u32(const void* p) {
    // generic LDS pointer: low 32 bits == wave-relative LDS byte offset
    return (uint32_t)(uintptr_t)p;
}

#define ASYNC_B128(ldsoff, voff, sbase)                                   \
    asm volatile("global_load_async_to_lds_b128 %0, %1, %2"               \
                 :: "v"(ldsoff), "v"(voff), "s"(sbase) : "memory")

// ---------------------------------------------------------------------------
// GEMM: C[M,N] = act(A[M,K] @ B[N,K]^T + bias), A/B bf16, strided batch.
// Block tile BM x BN, WROWS x WCOLS waves, wave tile (BM/WROWS) x (BN/WCOLS),
// K step 32, async double-buffered LDS staging.
// Requires: M % BM == 0, K % 32 == 0; N may be ragged (clamped staging,
// guarded epilogue).
// ---------------------------------------------------------------------------
template<int BM, int BN, int WROWS, int WCOLS,
         bool RELU, bool HAS_BIAS, bool STORE_F32, bool STORE_BF16>
__global__ __launch_bounds__(WROWS * WCOLS * 32)
void gemm_wmma_bf16(const uint16_t* __restrict__ Ag, long long strideAb, int lda,
                    const uint16_t* __restrict__ Bg, long long strideBb, int ldb,
                    const float* __restrict__ bias,
                    float* __restrict__ Cf, long long strideCf, int ldcf,
                    uint16_t* __restrict__ Cb, long long strideCb,
                    long long rsb, long long csb,
                    int M, int N, int K)
{
    constexpr int NW      = WROWS * WCOLS;
    constexpr int THREADS = NW * 32;
    constexpr int TM      = BM / (WROWS * 16);   // m-subtiles per wave
    constexpr int TN      = BN / (WCOLS * 16);   // n-subtiles per wave
    static_assert(BM * 4 == 2 * THREADS, "2 A-chunks per thread");
    static_assert(BN * 4 == 2 * THREADS, "2 B-chunks per thread");

    __shared__ __align__(16) uint16_t lA[2][BM * 32];
    __shared__ __align__(16) uint16_t lB[2][BN * 32];

    const int t    = threadIdx.x;
    const int lane = t & 31;
    const int wave = t >> 5;
    const int wrow = (wave / WCOLS) * (TM * 16);
    const int wcol = (wave % WCOLS) * (TN * 16);
    const int half = lane >> 4;
    const int ln   = lane & 15;

    const int rowBase = blockIdx.y * BM;
    const int colBase = blockIdx.x * BN;

    const unsigned long long Abase =
        (unsigned long long)(uintptr_t)(Ag + (long long)blockIdx.z * strideAb);
    const unsigned long long Bbase =
        (unsigned long long)(uintptr_t)(Bg + (long long)blockIdx.z * strideBb);

    // staging: tile row = 64B (32 bf16) = 4 x 16B chunks; thread t owns
    // chunks t and t+THREADS (rows t>>2 and THREADS/4 + (t>>2)).
    const int cbyte = (t & 3) * 16;
    const unsigned ldaB = (unsigned)lda * 2u;
    const unsigned ldbB = (unsigned)ldb * 2u;
    const int rStep = THREADS / 4;

    unsigned voffA0 = (unsigned)(rowBase + (t >> 2)) * ldaB + cbyte;
    unsigned voffA1 = (unsigned)(rowBase + rStep + (t >> 2)) * ldaB + cbyte;
    int nB0 = colBase + (t >> 2);          if (nB0 > N - 1) nB0 = N - 1;
    int nB1 = colBase + rStep + (t >> 2);  if (nB1 > N - 1) nB1 = N - 1;
    unsigned voffB0 = (unsigned)nB0 * ldbB + cbyte;
    unsigned voffB1 = (unsigned)nB1 * ldbB + cbyte;

    const uint32_t ldsAo = lds_off_u32(&lA[0][0]);
    const uint32_t ldsBo = lds_off_u32(&lB[0][0]);
    const uint32_t slot0 = (uint32_t)t * 16u;
    const uint32_t slot1 = (uint32_t)(t + THREADS) * 16u;

    auto stage = [&](int buf) {
        uint32_t oA = ldsAo + (uint32_t)buf * (BM * 64u);
        uint32_t oB = ldsBo + (uint32_t)buf * (BN * 64u);
        ASYNC_B128(oA + slot0, voffA0, Abase);
        ASYNC_B128(oA + slot1, voffA1, Abase);
        ASYNC_B128(oB + slot0, voffB0, Bbase);
        ASYNC_B128(oB + slot1, voffB1, Bbase);
        voffA0 += 64u; voffA1 += 64u; voffB0 += 64u; voffB1 += 64u;
    };

    v8f acc[TM][TN];
    #pragma unroll
    for (int i = 0; i < TM; ++i)
        #pragma unroll
        for (int j = 0; j < TN; ++j)
            acc[i][j] = (v8f){0.f,0.f,0.f,0.f,0.f,0.f,0.f,0.f};

    // 16-bit A fragment K-groups (ISA 7.12.2):
    // lanes 0-15: K 0-7 | 16-23 ; lanes 16-31: K 8-15 | 24-31
    const int g0 = half * 8;
    const int g1 = 16 + half * 8;

    stage(0);
    const int nIter = K >> 5;
    for (int it = 0; it < nIter; ++it) {
        const int cur = it & 1;
        if (it + 1 < nIter) {
            stage(cur ^ 1);
            asm volatile("s_wait_asynccnt 0x4" ::: "memory");
        } else {
            asm volatile("s_wait_asynccnt 0x0" ::: "memory");
        }
        __syncthreads();

        FragBF fa[TM], fb[TN];
        #pragma unroll
        for (int i = 0; i < TM; ++i) {
            int m = wrow + i * 16 + ln;
            fa[i].q[0] = *(const u32x4*)&lA[cur][m * 32 + g0];
            fa[i].q[1] = *(const u32x4*)&lA[cur][m * 32 + g1];
        }
        #pragma unroll
        for (int j = 0; j < TN; ++j) {
            int n = wcol + j * 16 + ln;
            fb[j].q[0] = *(const u32x4*)&lB[cur][n * 32 + g0];
            fb[j].q[1] = *(const u32x4*)&lB[cur][n * 32 + g1];
        }
        #pragma unroll
        for (int i = 0; i < TM; ++i)
            #pragma unroll
            for (int j = 0; j < TN; ++j)
                acc[i][j] = __builtin_amdgcn_wmma_f32_16x16x32_bf16(
                    false, fa[i].v, false, fb[j].v,
                    (short)0, acc[i][j], false, false);
        __syncthreads();
    }

    float* CfB = STORE_F32 ? Cf + (long long)blockIdx.z * strideCf : nullptr;
    uint16_t* CbB = STORE_BF16 ? Cb + (long long)blockIdx.z * strideCb : nullptr;

    #pragma unroll
    for (int i = 0; i < TM; ++i) {
        #pragma unroll
        for (int j = 0; j < TN; ++j) {
            int col = colBase + wcol + j * 16 + ln;
            if (col >= N) continue;
            float bv = HAS_BIAS ? bias[col] : 0.0f;
            #pragma unroll
            for (int r = 0; r < 8; ++r) {
                int row = rowBase + wrow + i * 16 + half * 8 + r;
                float v = acc[i][j][r] + bv;
                if (RELU) v = v > 0.f ? v : 0.f;
                if (STORE_F32)  CfB[(long long)row * ldcf + col] = v;
                if (STORE_BF16) CbB[(long long)row * rsb + (long long)col * csb] =
                                    f32_to_bf16_rne(v);
            }
        }
    }
}

// ---------------------------------------------------------------------------
// elementwise f32 -> bf16 (contiguous, n % 4 == 0)
__global__ __launch_bounds__(256)
void cvt_bf16_kernel(const float* __restrict__ in, uint16_t* __restrict__ out,
                     long long n)
{
    long long g = ((long long)blockIdx.x * 256 + threadIdx.x) * 4;
    if (g >= n) return;
    f32x4 v = *(const f32x4*)(in + g);
    u32x2 p;
    p[0] = (unsigned)f32_to_bf16_rne(v[0]) | ((unsigned)f32_to_bf16_rne(v[1]) << 16);
    p[1] = (unsigned)f32_to_bf16_rne(v[2]) | ((unsigned)f32_to_bf16_rne(v[3]) << 16);
    *(u32x2*)(out + g) = p;
}

// f32 rows of EE -> bf16 rows of (EE+HH) (fills cat[:, 0:E])
__global__ __launch_bounds__(256)
void cvt_bf16_rows_kernel(const float* __restrict__ in, uint16_t* __restrict__ out,
                          int rows)
{
    long long g = (long long)blockIdx.x * 256 + threadIdx.x;   // groups of 4
    long long tot = (long long)rows * (EE / 4);
    if (g >= tot) return;
    long long m = g / (EE / 4);
    int c4 = (int)(g % (EE / 4)) * 4;
    f32x4 v = *(const f32x4*)(in + m * EE + c4);
    u32x2 p;
    p[0] = (unsigned)f32_to_bf16_rne(v[0]) | ((unsigned)f32_to_bf16_rne(v[1]) << 16);
    p[1] = (unsigned)f32_to_bf16_rne(v[2]) | ((unsigned)f32_to_bf16_rne(v[3]) << 16);
    *(u32x2*)(out + m * (EE + HH) + c4) = p;
}

// ---------------------------------------------------------------------------
// weight[b,a] = (x·muw + mub) + exp(0.5*(x·lvw + lvb)) * eps   (wave per row)
__global__ __launch_bounds__(256)
void comm_weight_kernel(const float* __restrict__ inputs,
                        const float* __restrict__ muw, const float* __restrict__ mub,
                        const float* __restrict__ lvw, const float* __restrict__ lvb,
                        const float* __restrict__ eps,
                        float* __restrict__ weight, int rows)
{
    int wave = threadIdx.x >> 5, lane = threadIdx.x & 31;
    int row = blockIdx.x * 8 + wave;
    if (row >= rows) return;
    const float* x = inputs + (long long)row * EE;
    float s0 = 0.f, s1 = 0.f;
    #pragma unroll
    for (int c = lane; c < EE; c += 32) {
        float xv = x[c];
        s0 += xv * muw[c];
        s1 += xv * lvw[c];
    }
    #pragma unroll
    for (int o = 16; o > 0; o >>= 1) {
        s0 += __shfl_down(s0, o, 32);
        s1 += __shfl_down(s1, o, 32);
    }
    if (lane == 0) {
        weight[row] = (s0 + mub[0]) + __expf(0.5f * (s1 + lvb[0])) * eps[row];
    }
}

// ---------------------------------------------------------------------------
// vm[b,i,:] (bf16): zero KSEL smallest of weight[b,:]*visible[b,i,:]
// (tie-break: smaller index), then += identity. One wave per row.
__global__ __launch_bounds__(256)
void vm_kernel(const float* __restrict__ weight,
               const float* __restrict__ visible,
               uint16_t* __restrict__ vm, int rows)
{
    int wave = threadIdx.x >> 5, lane = threadIdx.x & 31;
    int row = blockIdx.x * 8 + wave;
    if (row >= rows) return;
    int b = row / AA, i = row % AA;
    const float* visRow = visible + (long long)row * AA;
    const float* wRow   = weight + (long long)b * AA;

    float v0 = wRow[lane]      * visRow[lane];
    float v1 = wRow[lane + 32] * visRow[lane + 32];
    bool rm0 = false, rm1 = false;

    for (int it = 0; it < KSEL; ++it) {
        float cv = 3.4e38f; int ci = 1 << 30;
        if (!rm0) { cv = v0; ci = lane; }
        if (!rm1 && (v1 < cv || (v1 == cv && (lane + 32) < ci))) { cv = v1; ci = lane + 32; }
        #pragma unroll
        for (int o = 16; o > 0; o >>= 1) {
            float ov = __shfl_xor(cv, o, 32);
            int   oi = __shfl_xor(ci, o, 32);
            if (ov < cv || (ov == cv && oi < ci)) { cv = ov; ci = oi; }
        }
        if (ci == lane)           rm0 = true;
        else if (ci == lane + 32) rm1 = true;
    }
    float o0 = rm0 ? 0.f : visRow[lane];
    float o1 = rm1 ? 0.f : visRow[lane + 32];
    if (lane == i)      o0 += 1.0f;
    if (lane + 32 == i) o1 += 1.0f;
    vm[(long long)row * AA + lane]      = f32_to_bf16_rne(o0);
    vm[(long long)row * AA + lane + 32] = f32_to_bf16_rne(o1);
}

// ---------------------------------------------------------------------------
// GRU combine; dual output: f32 (d_out) + bf16 (feeds next GEMM)
__global__ __launch_bounds__(256)
void gru_combine_kernel(const float* __restrict__ gi, const float* __restrict__ gh,
                        const float* __restrict__ h,
                        float* __restrict__ outf, uint16_t* __restrict__ outb, int M)
{
    long long idx = (long long)blockIdx.x * blockDim.x + threadIdx.x;
    long long tot = (long long)M * HH;
    if (idx >= tot) return;
    long long m = idx / HH; int c = (int)(idx % HH);
    const float* gim = gi + m * (3 * HH);
    const float* ghm = gh + m * (3 * HH);
    float r = 1.0f / (1.0f + __expf(-(gim[c] + ghm[c])));
    float z = 1.0f / (1.0f + __expf(-(gim[HH + c] + ghm[HH + c])));
    float n = tanhf(gim[2 * HH + c] + r * ghm[2 * HH + c]);
    float v = (1.0f - z) * n + z * h[idx];
    outf[idx] = v;
    outb[idx] = f32_to_bf16_rne(v);
}

// ---------------------------------------------------------------------------
extern "C" void kernel_launch(void* const* d_in, const int* in_sizes, int n_in,
                              void* d_out, int out_size, void* d_ws, size_t ws_size,
                              hipStream_t stream)
{
    (void)in_sizes; (void)n_in; (void)out_size; (void)ws_size;

    const float* inputs   = (const float*)d_in[0];
    const float* visible  = (const float*)d_in[1];
    const float* hstate   = (const float*)d_in[2];
    const float* hstate2  = (const float*)d_in[3];
    const float* eps      = (const float*)d_in[4];
    const float* h2mu_w   = (const float*)d_in[5];
    const float* h2mu_b   = (const float*)d_in[6];
    const float* h2lv_w   = (const float*)d_in[7];
    const float* h2lv_b   = (const float*)d_in[8];
    const float* fcn_w    = (const float*)d_in[9];
    const float* fcn_b    = (const float*)d_in[10];
    const float* rnn_wih  = (const float*)d_in[11];
    const float* rnn_whh  = (const float*)d_in[12];
    const float* rnn_bih  = (const float*)d_in[13];
    const float* rnn_bhh  = (const float*)d_in[14];
    const float* fc2n_w   = (const float*)d_in[15];
    const float* fc2n_b   = (const float*)d_in[16];
    const float* fc1_w    = (const float*)d_in[17];
    const float* fc1_b    = (const float*)d_in[18];
    const float* rnn2_wih = (const float*)d_in[19];
    const float* rnn2_whh = (const float*)d_in[20];
    const float* rnn2_bih = (const float*)d_in[21];
    const float* rnn2_bhh = (const float*)d_in[22];
    const float* fc2_w    = (const float*)d_in[23];
    const float* fc2_b    = (const float*)d_in[24];

    const int M = BB * AA;   // 32768

    float* q_out   = (float*)d_out;
    float* hh_out  = q_out + (long long)M * NAA;
    float* hhn_out = hh_out + (long long)M * HH;

    // ---- workspace carving (256B aligned) ----
    char* cur = (char*)d_ws;
    auto alloc = [&](long long bytes) {
        char* p = cur;
        cur += (bytes + 255) & ~255LL;
        return (void*)p;
    };
    float*    weight  = (float*)   alloc((long long)M * 4);
    uint16_t* vm_bf   = (uint16_t*)alloc((long long)BB * AA * AA * 2);
    uint16_t* PT_bf   = (uint16_t*)alloc((long long)BB * HH * AA * 2);  // [b][h][j]
    uint16_t* neigh_bf= (uint16_t*)alloc((long long)M * HH * 2);
    uint16_t* cat_bf  = (uint16_t*)alloc((long long)M * (EE + HH) * 2); // [m][512]
    uint16_t* h1_bf   = (uint16_t*)alloc((long long)M * HH * 2);
    uint16_t* h2_bf   = (uint16_t*)alloc((long long)M * HH * 2);
    uint16_t* hhn_bf  = (uint16_t*)alloc((long long)M * HH * 2);
    float*    gi      = (float*)   alloc((long long)M * 3 * HH * 4);
    float*    gh      = (float*)   alloc((long long)M * 3 * HH * 4);
    uint16_t* fcnw_bf = (uint16_t*)alloc((long long)HH * AA * EE * 2);
    uint16_t* wih_bf  = (uint16_t*)alloc((long long)3 * HH * HH * 2);
    uint16_t* whh_bf  = (uint16_t*)alloc((long long)3 * HH * HH * 2);
    uint16_t* wih2_bf = (uint16_t*)alloc((long long)3 * HH * HH * 2);
    uint16_t* whh2_bf = (uint16_t*)alloc((long long)3 * HH * HH * 2);
    uint16_t* fc2n_bf = (uint16_t*)alloc((long long)HH * HH * 2);
    uint16_t* fc1_bf  = (uint16_t*)alloc((long long)HH * (EE + HH) * 2);
    uint16_t* fc2_bf  = (uint16_t*)alloc((long long)NAA * HH * 2);
    uint16_t* x_bf    = PT_bf;    // PT dead after neigh GEMM
    uint16_t* hh_bf   = hhn_bf;   // hhn_bf dead after n3 GEMM

    auto cvt = [&](const float* src, uint16_t* dst, long long n) {
        cvt_bf16_kernel<<<(unsigned)((n / 4 + 255) / 256), 256, 0, stream>>>(src, dst, n);
    };

    // ---- operand conversions (memory-bound, one pass each) ----
    cvt_bf16_rows_kernel<<<(unsigned)(((long long)M * (EE/4) + 255) / 256), 256, 0, stream>>>(
        inputs, cat_bf, M);                              // cat[:, 0:E]
    cvt(hstate,   h1_bf,  (long long)M * HH);
    cvt(hstate2,  h2_bf,  (long long)M * HH);
    cvt(fcn_w,    fcnw_bf,(long long)HH * AA * EE);
    cvt(rnn_wih,  wih_bf, (long long)3 * HH * HH);
    cvt(rnn_whh,  whh_bf, (long long)3 * HH * HH);
    cvt(rnn2_wih, wih2_bf,(long long)3 * HH * HH);
    cvt(rnn2_whh, whh2_bf,(long long)3 * HH * HH);
    cvt(fc2n_w,   fc2n_bf,(long long)HH * HH);
    cvt(fc1_w,    fc1_bf, (long long)HH * (EE + HH));
    cvt(fc2_w,    fc2_bf, (long long)NAA * HH);

    // 1) comm weights ; 2) masked adjacency (bf16)
    comm_weight_kernel<<<M / 8, 256, 0, stream>>>(inputs, h2mu_w, h2mu_b,
                                                  h2lv_w, h2lv_b, eps, weight, M);
    vm_kernel<<<M / 8, 256, 0, stream>>>(weight, visible, vm_bf, M);

    // 3) PT[b][h][j] = sum_f inputs[b,j,f] * W3[h,j,f]   (batched over j, 64x64)
    gemm_wmma_bf16<64, 64, 2, 2, false, false, false, true>
        <<<dim3(HH/64, BB/64, AA), 128, 0, stream>>>(
        cat_bf, (long long)(EE + HH), AA * (EE + HH),
        fcnw_bf, (long long)EE, AA * EE,
        nullptr,
        nullptr, 0LL, 0,
        PT_bf, 1LL, (long long)HH * AA, (long long)AA,
        BB, HH, EE);

    // 4) neigh = relu(vm @ PT^T + fcn_b)  (batched over b, 64x64)
    gemm_wmma_bf16<64, 64, 2, 2, true, true, false, true>
        <<<dim3(HH/64, 1, BB), 128, 0, stream>>>(
        vm_bf, (long long)AA * AA, AA,
        PT_bf, (long long)HH * AA, AA,
        fcn_b,
        nullptr, 0LL, 0,
        neigh_bf, (long long)AA * HH, (long long)HH, 1LL,
        AA, HH, AA);

    // 5/6) neighbor GRU GEMMs (f32 gates), 128x128
    gemm_wmma_bf16<128, 128, 4, 2, false, true, true, false>
        <<<dim3(3*HH/128, M/128, 1), 256, 0, stream>>>(
        neigh_bf, 0LL, HH, wih_bf, 0LL, HH, rnn_bih,
        gi, 0LL, 3 * HH, nullptr, 0LL, 0LL, 0LL, M, 3 * HH, HH);
    gemm_wmma_bf16<128, 128, 4, 2, false, true, true, false>
        <<<dim3(3*HH/128, M/128, 1), 256, 0, stream>>>(
        h2_bf, 0LL, HH, whh_bf, 0LL, HH, rnn_bhh,
        gh, 0LL, 3 * HH, nullptr, 0LL, 0LL, 0LL, M, 3 * HH, HH);

    // 7) hh_neigh (f32 to d_out + bf16)
    gru_combine_kernel<<<(unsigned)(((long long)M * HH + 255) / 256), 256, 0, stream>>>(
        gi, gh, hstate2, hhn_out, hhn_bf, M);

    // 8) n3 -> cat[:, E:] (bf16, row stride 512), 128x128
    gemm_wmma_bf16<128, 128, 4, 2, false, true, false, true>
        <<<dim3(HH/128, M/128, 1), 256, 0, stream>>>(
        hhn_bf, 0LL, HH, fc2n_bf, 0LL, HH, fc2n_b,
        nullptr, 0LL, 0,
        cat_bf + EE, 0LL, (long long)(EE + HH), 1LL,
        M, HH, HH);

    // 10) x = relu(cat @ fc1^T + b), K = 512, 128x128
    gemm_wmma_bf16<128, 128, 4, 2, true, true, false, true>
        <<<dim3(HH/128, M/128, 1), 256, 0, stream>>>(
        cat_bf, 0LL, EE + HH, fc1_bf, 0LL, EE + HH, fc1_b,
        nullptr, 0LL, 0,
        x_bf, 0LL, (long long)HH, 1LL,
        M, HH, EE + HH);

    // 11/12) main GRU GEMMs, 128x128
    gemm_wmma_bf16<128, 128, 4, 2, false, true, true, false>
        <<<dim3(3*HH/128, M/128, 1), 256, 0, stream>>>(
        x_bf, 0LL, HH, wih2_bf, 0LL, HH, rnn2_bih,
        gi, 0LL, 3 * HH, nullptr, 0LL, 0LL, 0LL, M, 3 * HH, HH);
    gemm_wmma_bf16<128, 128, 4, 2, false, true, true, false>
        <<<dim3(3*HH/128, M/128, 1), 256, 0, stream>>>(
        h1_bf, 0LL, HH, whh2_bf, 0LL, HH, rnn2_bhh,
        gh, 0LL, 3 * HH, nullptr, 0LL, 0LL, 0LL, M, 3 * HH, HH);

    // 13) hh (f32 to d_out + bf16)
    gru_combine_kernel<<<(unsigned)(((long long)M * HH + 255) / 256), 256, 0, stream>>>(
        gi, gh, hstate, hh_out, hh_bf, M);

    // 14) q = hh @ fc2^T + b  (N=32: staging clamps rows, epilogue guards cols)
    gemm_wmma_bf16<128, 128, 4, 2, false, true, true, false>
        <<<dim3(1, M/128, 1), 256, 0, stream>>>(
        hh_bf, 0LL, HH, fc2_bf, 0LL, HH, fc2_b,
        q_out, 0LL, NAA, nullptr, 0LL, 0LL, 0LL, M, NAA, HH);
}
